// DConvRNN_38568806318198
// MI455X (gfx1250) — compile-verified, matrix operand order checked by reading, not run on hardware
//
#include <hip/hip_runtime.h>
#include <math.h>

typedef __attribute__((ext_vector_type(16))) _Float16 v16h;
typedef __attribute__((ext_vector_type(8)))  float    v8f;

#define NN      50000      // nodes
#define NE      800000     // edges
#define FEAT    16
#define HID     64
#define KCHEB   32
#define NCLS    8
#define FIN     80                     // FEAT + HID (weight fin-stride region)
#define W_K_STRIDE (FIN * HID)         // 5120 floats per k
#define W_D_STRIDE (KCHEB * FIN * HID) // 163840 floats per direction

// ---------- utility kernels ----------
__global__ void zero_f32(float* p, int n) {
    int i = blockIdx.x * blockDim.x + threadIdx.x;
    if (i < n) p[i] = 0.f;
}

__global__ void neg_inplace(float* a, float* b, int n) {
    int i = blockIdx.x * blockDim.x + threadIdx.x;
    if (i < n) { a[i] = -a[i]; b[i] = -b[i]; }
}

__global__ void copy2(const float* __restrict__ x, float* a, float* b, int n) {
    int i = blockIdx.x * blockDim.x + threadIdx.x;
    if (i < n) { float v = x[i]; a[i] = v; b[i] = v; }
}

// ---------- degree / normalization ----------
__global__ void deg_kernel(const int* __restrict__ src, const int* __restrict__ dst,
                           float* deg_out, float* deg_in) {
    int i = blockIdx.x * blockDim.x + threadIdx.x;
    if (i < NE) {
        atomicAdd(&deg_out[src[i]], 1.0f);
    } else if (i < 2 * NE) {
        atomicAdd(&deg_in[dst[i - NE]], 1.0f);
    }
}

__global__ void inv_kernel(float* d, int n) {
    int i = blockIdx.x * blockDim.x + threadIdx.x;
    if (i < n) { float v = d[i]; d[i] = (v > 0.f) ? (1.f / v) : 0.f; }
}

// loop-invariant per-edge normalization weights (hoisted out of the 31-step chain)
__global__ void edge_norm(const int* __restrict__ src, const int* __restrict__ dst,
                          const float* __restrict__ invdo, const float* __restrict__ invdi,
                          float* norm_o, float* norm_i) {
    int e = blockIdx.x * blockDim.x + threadIdx.x;
    if (e < NE) {
        norm_o[e] = invdo[src[e]];
        norm_i[e] = invdi[dst[e]];
    }
}

// ---------- sparse diffusion: T_next += scale * norm[e] * gather(T_cur) ----------
// forward:  out[dst[e]] += T_o[src[e]] * norm_o[e]
// reverse:  out[src[e]] += T_i[dst[e]] * norm_i[e]
// one thread per (edge, feature, direction); half-wave-coalesced on features
__global__ void edge_prop(const float* __restrict__ go, float* so,
                          const float* __restrict__ gi, float* si,
                          const int* __restrict__ src, const int* __restrict__ dst,
                          const float* __restrict__ no, const float* __restrict__ ni,
                          float scale) {
    int i = blockIdx.x * blockDim.x + threadIdx.x;
    const int tot = NE * FEAT;  // 12.8M
    if (i < tot) {
        int e = i >> 4, f = i & 15;
        float v = go[src[e] * FEAT + f] * (no[e] * scale);
        atomicAdd(&so[dst[e] * FEAT + f], v);
    } else if (i < 2 * tot) {
        int j = i - tot;
        int e = j >> 4, f = j & 15;
        float v = gi[dst[e] * FEAT + f] * (ni[e] * scale);
        atomicAdd(&si[src[e] * FEAT + f], v);
    }
}

// ---------- WMMA accumulate: Hacc[16 rows x 128] += [Tk_o | Tk_i](16x32) @ Wcat(32x128) ----
// 4 waves/block: wave = gate(2) x colpair(2); each wave does TWO 16x16 output tiles,
// sharing one A fragment (loaded as 4x b128) across two back-to-back WMMAs.
// Contraction index kk: 0..15 -> Tk_o / W[0][k], 16..31 -> Tk_i / W[1][k].
__global__ __launch_bounds__(128) void wmma_acc(const float* __restrict__ Ao,
                                                const float* __restrict__ Ai,
                                                const float* __restrict__ Wz,
                                                const float* __restrict__ Wh,
                                                float* Hacc, int k) {
    const int lane  = threadIdx.x & 31;
    const int wave  = threadIdx.x >> 5;   // 0..3
    const int gate  = wave >> 1;          // 0 = z, 1 = h
    const int cp    = wave & 1;           // column pair: cols [cp*32, cp*32+32)
    const int row0  = blockIdx.x * 16;    // 3125 * 16 == 50000 exactly
    const int m     = lane & 15;
    const int half  = lane >> 4;          // lanes 0-15: K 0-7/16-23, lanes 16-31: K 8-15/24-31
    const int kbase = half * 8;
    const float* Wg = gate ? Wh : Wz;

    // ---- A fragment (shared by both column tiles): 32B-aligned vector loads
    const float4* pAo = (const float4*)(Ao + (row0 + m) * FEAT + kbase);
    const float4* pAi = (const float4*)(Ai + (row0 + m) * FEAT + kbase);
    float4 a0 = pAo[0], a1 = pAo[1];     // Tk_o feats kbase..kbase+7
    float4 a2 = pAi[0], a3 = pAi[1];     // Tk_i feats kbase..kbase+7
    v16h a;
    a[0]  = (_Float16)a0.x; a[1]  = (_Float16)a0.y; a[2]  = (_Float16)a0.z; a[3]  = (_Float16)a0.w;
    a[4]  = (_Float16)a1.x; a[5]  = (_Float16)a1.y; a[6]  = (_Float16)a1.z; a[7]  = (_Float16)a1.w;
    a[8]  = (_Float16)a2.x; a[9]  = (_Float16)a2.y; a[10] = (_Float16)a2.z; a[11] = (_Float16)a2.w;
    a[12] = (_Float16)a3.x; a[13] = (_Float16)a3.y; a[14] = (_Float16)a3.z; a[15] = (_Float16)a3.w;

    const float* Wk = Wg + (size_t)k * W_K_STRIDE;

#pragma unroll
    for (int t = 0; t < 2; ++t) {
        const int col0 = cp * 32 + t * 16;
        // ---- B fragment: Wcat[kk][col0+m]; j<8 -> dir 0 (fin=kbase+j), j>=8 -> dir 1
        v16h b;
#pragma unroll
        for (int j = 0; j < 8; ++j) {
            float b0 = Wk[(kbase + j) * HID + col0 + m];                 // W[0][k]
            float b1 = Wk[W_D_STRIDE + (kbase + j) * HID + col0 + m];    // W[1][k]
            b[j]     = (_Float16)b0;
            b[j + 8] = (_Float16)b1;
        }
        // ---- C tile (FP32 accumulate in Hacc)
        v8f c;
#pragma unroll
        for (int j = 0; j < 8; ++j) {
            int mm = j + 8 * half;
            c[j] = Hacc[(row0 + mm) * 128 + gate * 64 + col0 + m];
        }
        c = __builtin_amdgcn_wmma_f32_16x16x32_f16(false, a, false, b, (short)0, c,
                                                   false, false);
#pragma unroll
        for (int j = 0; j < 8; ++j) {
            int mm = j + 8 * half;
            Hacc[(row0 + mm) * 128 + gate * 64 + col0 + m] = c[j];
        }
    }
}

// ---------- GRU epilogue + classifier: out = relu((1-sigmoid(hz+bz))*tanh(hh+bh)) @ lin_w + lin_b
__global__ void finalize(const float* __restrict__ Hacc,
                         const float* __restrict__ bz, const float* __restrict__ bh,
                         const float* __restrict__ lin_w, const float* __restrict__ lin_b,
                         float* out) {
    int n = blockIdx.x * blockDim.x + threadIdx.x;
    if (n >= NN) return;
    const float* hz = Hacc + (size_t)n * 128;
    const float* hh = hz + 64;
    float acc[NCLS];
#pragma unroll
    for (int c = 0; c < NCLS; ++c) acc[c] = lin_b[c];
    for (int j = 0; j < HID; ++j) {
        float z  = 1.f / (1.f + __expf(-(hz[j] + bz[j])));
        float ht = tanhf(hh[j] + bh[j]);
        float h  = (1.f - z) * ht;
        float r  = (h > 0.f) ? h : 0.f;
#pragma unroll
        for (int c = 0; c < NCLS; ++c) acc[c] += r * lin_w[j * NCLS + c];
    }
#pragma unroll
    for (int c = 0; c < NCLS; ++c) out[n * NCLS + c] = acc[c];
}

extern "C" void kernel_launch(void* const* d_in, const int* in_sizes, int n_in,
                              void* d_out, int out_size, void* d_ws, size_t ws_size,
                              hipStream_t stream) {
    const float* x     = (const float*)d_in[0];
    const int*   ei    = (const int*)d_in[1];
    const int*   src   = ei;
    const int*   dst   = ei + NE;
    const float* Wz    = (const float*)d_in[2];
    const float* bz    = (const float*)d_in[3];
    // d_in[4] (W_r) / d_in[5] (b_r) are mathematically unused: H0 == 0
    const float* Wh    = (const float*)d_in[6];
    const float* bh    = (const float*)d_in[7];
    const float* lin_w = (const float*)d_in[8];
    const float* lin_b = (const float*)d_in[9];
    float*       out   = (float*)d_out;

    // workspace layout (floats): ~45 MB total
    float* ws      = (float*)d_ws;
    float* deg_out = ws;                      // NN
    float* deg_in  = deg_out + NN;            // NN (contiguous with deg_out)
    float* norm_o  = deg_in + NN;             // NE
    float* norm_i  = norm_o + NE;             // NE
    float* To0     = norm_i + NE;             // NN*FEAT
    float* To1     = To0 + NN * FEAT;
    float* Ti0     = To1 + NN * FEAT;
    float* Ti1     = Ti0 + NN * FEAT;
    float* Hacc    = Ti1 + NN * FEAT;         // NN*128

    const int NT  = NN * FEAT;
    const int TPB = 256;
    const int gT  = (NT + TPB - 1) / TPB;
    const int gE  = (2 * NE * FEAT + TPB - 1) / TPB;

    // degrees -> inverse -> per-edge norms (all loop-invariant)
    zero_f32<<<(2 * NN + TPB - 1) / TPB, TPB, 0, stream>>>(deg_out, 2 * NN);
    deg_kernel<<<(2 * NE + TPB - 1) / TPB, TPB, 0, stream>>>(src, dst, deg_out, deg_in);
    inv_kernel<<<(2 * NN + TPB - 1) / TPB, TPB, 0, stream>>>(deg_out, 2 * NN);
    edge_norm<<<(NE + TPB - 1) / TPB, TPB, 0, stream>>>(src, dst, deg_out, deg_in,
                                                        norm_o, norm_i);

    // accumulator and T0 = Xc (only x columns nonzero)
    zero_f32<<<(NN * 128 + TPB - 1) / TPB, TPB, 0, stream>>>(Hacc, NN * 128);
    copy2<<<gT, TPB, 0, stream>>>(x, To0, Ti0, NT);

    float *To_prev = To0, *To_cur = To1, *Ti_prev = Ti0, *Ti_cur = Ti1;

    // k = 0 : identity term, both directions use Xc
    wmma_acc<<<NN / 16, 128, 0, stream>>>(To_prev, Ti_prev, Wz, Wh, Hacc, 0);

    // k = 1 : T1 = prop(T0)
    zero_f32<<<gT, TPB, 0, stream>>>(To_cur, NT);
    zero_f32<<<gT, TPB, 0, stream>>>(Ti_cur, NT);
    edge_prop<<<gE, TPB, 0, stream>>>(To_prev, To_cur, Ti_prev, Ti_cur,
                                      src, dst, norm_o, norm_i, 1.0f);
    wmma_acc<<<NN / 16, 128, 0, stream>>>(To_cur, Ti_cur, Wz, Wh, Hacc, 1);

    // k >= 2 : T_next = 2*prop(T_cur) - T_prev  (in-place into prev buffer, then rotate)
    for (int k = 2; k < KCHEB; ++k) {
        neg_inplace<<<gT, TPB, 0, stream>>>(To_prev, Ti_prev, NT);
        edge_prop<<<gE, TPB, 0, stream>>>(To_cur, To_prev, Ti_cur, Ti_prev,
                                          src, dst, norm_o, norm_i, 2.0f);
        float* t;
        t = To_prev; To_prev = To_cur; To_cur = t;
        t = Ti_prev; Ti_prev = Ti_cur; Ti_cur = t;
        wmma_acc<<<NN / 16, 128, 0, stream>>>(To_cur, Ti_cur, Wz, Wh, Hacc, k);
    }

    finalize<<<(NN + TPB - 1) / TPB, TPB, 0, stream>>>(Hacc, bz, bh, lin_w, lin_b, out);
}
